// FlowAttention_40072044872138
// MI455X (gfx1250) — matrix operand order repaired and checked by reading, not compiled, not run on hardware
//
#include <hip/hip_runtime.h>
#include <hip/hip_bf16.h>

typedef __bf16 bf16;
typedef __attribute__((ext_vector_type(16))) __bf16 v16bf;
typedef __attribute__((ext_vector_type(8)))  __bf16 v8bf;
typedef __attribute__((ext_vector_type(8)))  float  v8f;

#define T_SEQ 2048
#define BSZ   2
#define EMB   1024
#define NH    16
#define DH    64
#define BH    (BSZ * NH)      /* 32  */
#define ROWS  (T_SEQ * BSZ)   /* 4096 */
#define SCALING 0.125f        /* 64^-0.5 */

// ---------------------------------------------------------------------------
// WMMA fragment helpers (CDNA5 16x16x32 bf16, wave32)
// A (16x32): lane holds row (lane&15), K = (lane>>4)*16 + 0..15 (contiguous)
// B (32x16): lane holds col (lane&15), K = (lane>>4)*16 + 0..15 (contiguous)
// C (16x16 f32): lane holds col (lane&15), rows M = v + (lane>>4)*8
// ---------------------------------------------------------------------------
__device__ __forceinline__ v16bf a_frag_from_f32(const float* rowp, int kb) {
  float f[16];
  *(float4*)(f + 0)  = *(const float4*)(rowp + kb);
  *(float4*)(f + 4)  = *(const float4*)(rowp + kb + 4);
  *(float4*)(f + 8)  = *(const float4*)(rowp + kb + 8);
  *(float4*)(f + 12) = *(const float4*)(rowp + kb + 12);
  v16bf a;
#pragma unroll
  for (int i = 0; i < 16; ++i) a[i] = (__bf16)f[i];
  return a;
}

__device__ __forceinline__ v16bf a_frag_from_bf16(const bf16* rowp, int kb) {
  union { v16bf v; v8bf h[2]; } u;
  u.h[0] = *(const v8bf*)(rowp + kb);
  u.h[1] = *(const v8bf*)(rowp + kb + 8);
  return u.v;
}

__device__ __forceinline__ v8f wmma_bf16(v16bf a, v16bf b, v8f c) {
  return __builtin_amdgcn_wmma_f32_16x16x32_bf16(false, a, false, b,
                                                 (short)0, c, false, false);
}

// ---------------------------------------------------------------------------
// Kernel 0: one-time f32 -> bf16 down-convert (query + the 4 weight matrices)
// ---------------------------------------------------------------------------
__global__ void __launch_bounds__(256, 4)
cvt_bf16_kernel(const float* __restrict__ src, bf16* __restrict__ dst, int n) {
  int i = (blockIdx.x * blockDim.x + threadIdx.x) * 8;
  if (i >= n) return;
  float4 x0 = *(const float4*)(src + i);
  float4 x1 = *(const float4*)(src + i + 4);
  v8bf o;
  o[0] = (__bf16)x0.x; o[1] = (__bf16)x0.y; o[2] = (__bf16)x0.z; o[3] = (__bf16)x0.w;
  o[4] = (__bf16)x1.x; o[5] = (__bf16)x1.y; o[6] = (__bf16)x1.z; o[7] = (__bf16)x1.w;
  *(v8bf*)(dst + i) = o;
}

// ---------------------------------------------------------------------------
// Kernel 1: fused QKV projection, 4x4 register-blocked (64x64 per wave).
// Single base pointer per operand; mi/ni/e0 folded into constant instruction
// offsets (<100KB, fits 24-bit IOFFSET) -> no per-iteration address VALU.
// __launch_bounds__(32,1): full VGPR budget, no accumulator spills.
// ---------------------------------------------------------------------------
__global__ void __launch_bounds__(32, 1)
qkv_proj_kernel(const bf16* __restrict__ Xb,
                const bf16* __restrict__ Wqb,
                const bf16* __restrict__ Wkb,
                const bf16* __restrict__ Wvb,
                const float* __restrict__ bq,
                const float* __restrict__ bk,
                const float* __restrict__ bv,
                bf16* __restrict__ Qh,
                bf16* __restrict__ Kh,
                bf16* __restrict__ Vt) {
  const int lane  = threadIdx.x;
  const int r0    = blockIdx.x * 64;
  const int o0    = blockIdx.y * 64;
  const int which = blockIdx.z;

  const bf16*  Wb   = (which == 0) ? Wqb : (which == 1 ? Wkb : Wvb);
  const float* bias = (which == 0) ? bq : (which == 1 ? bk : bv);
  const float  scale = (which == 0) ? SCALING : 1.0f;

  const int n   = lane & 15;
  const int hi  = lane >> 4;
  const int kbA = hi * 16;
  const int kbB = hi * 16;

  const bf16* xbase = Xb + (size_t)(r0 + n) * EMB;          // + mi*16*EMB + e0
  const bf16* wbase = Wb + (size_t)(o0 + n) * EMB + kbB;    // + ni*16*EMB + e0

  v8f acc[4][4] = {};
  for (int e0 = 0; e0 < EMB; e0 += 32) {
    __builtin_prefetch(xbase + e0 + 256, 0, 3);
    __builtin_prefetch(wbase + e0 + 256, 0, 3);
    v16bf a[4], b[4];
#pragma unroll
    for (int mi = 0; mi < 4; ++mi)
      a[mi] = a_frag_from_bf16(xbase + mi * 16 * EMB + e0, kbA);
#pragma unroll
    for (int ni = 0; ni < 4; ++ni)
      b[ni] = *(const v16bf*)(wbase + ni * 16 * EMB + e0);
#pragma unroll
    for (int mi = 0; mi < 4; ++mi)
#pragma unroll
      for (int ni = 0; ni < 4; ++ni)
        acc[mi][ni] = wmma_bf16(a[mi], b[ni], acc[mi][ni]);
  }

  // epilogue: bias + scale + per-head scatter (branch hoisted out of loops)
  if (which == 2) {
#pragma unroll
    for (int mi = 0; mi < 4; ++mi)
#pragma unroll
      for (int ni = 0; ni < 4; ++ni) {
        const int o = o0 + ni * 16 + n;
        const int h = o >> 6, d = o & 63;
#pragma unroll
        for (int v = 0; v < 8; ++v) {
          int r = r0 + mi * 16 + v + hi * 8;
          int t = r >> 1, bb = r & 1;
          float val = acc[mi][ni][v] + bias[(size_t)t * EMB + o];
          Vt[(size_t)(bb * NH + h) * DH * T_SEQ + (size_t)d * T_SEQ + t] = (bf16)val;
        }
      }
  } else {
    bf16* dst = (which == 0) ? Qh : Kh;
#pragma unroll
    for (int mi = 0; mi < 4; ++mi)
#pragma unroll
      for (int ni = 0; ni < 4; ++ni) {
        const int o = o0 + ni * 16 + n;
        const int h = o >> 6, d = o & 63;
#pragma unroll
        for (int v = 0; v < 8; ++v) {
          int r = r0 + mi * 16 + v + hi * 8;
          int t = r >> 1, bb = r & 1;
          float val = (acc[mi][ni][v] + bias[(size_t)t * EMB + o]) * scale;
          dst[(size_t)(bb * NH + h) * T_SEQ * DH + (size_t)t * DH + d] = (bf16)val;
        }
      }
  }
}

// ---------------------------------------------------------------------------
// Kernel 2: attention for one (head, 16-row tile). 4 waves / 128 threads.
// LDS: 16 x 2048 f32 score band (128 KB, CDNA5 320KB WGP LDS).
// ---------------------------------------------------------------------------
__global__ void __launch_bounds__(128, 1)
attention_kernel(const bf16* __restrict__ Qh,
                 const bf16* __restrict__ Kh,
                 const bf16* __restrict__ Vt,
                 float* __restrict__ AW,
                 bf16* __restrict__ Ctx) {
  extern __shared__ float sm[];          // [16][T_SEQ]
  float* redmax = sm + 16 * T_SEQ;       // [16*8]
  float* redsum = redmax + 128;          // [16*8]
  float* rs     = redsum + 128;          // [16]

  const int tid  = threadIdx.x;
  const int lane = tid & 31;
  const int wave = tid >> 5;
  const int t0 = blockIdx.x * 16;
  const int bh = blockIdx.y;

  const int n   = lane & 15;
  const int hi  = lane >> 4;
  const int kbA = hi * 16;
  const int kbB = hi * 16;

  // ---- Phase 1: scores (Q frags pinned in VGPRs across all column tiles)
  const bf16* qrow = Qh + (size_t)bh * T_SEQ * DH + (size_t)(t0 + n) * DH;
  v16bf qa0 = a_frag_from_bf16(qrow, kbA);        // d = 0..31
  v16bf qa1 = a_frag_from_bf16(qrow + 32, kbA);   // d = 32..63
  for (int st = wave; st < T_SEQ / 16; st += 4) {
    int s0 = st * 16;
    const bf16* krow = Kh + (size_t)bh * T_SEQ * DH + (size_t)(s0 + n) * DH;
    __builtin_prefetch(krow + 64 * DH, 0, 3);
    v16bf b0 = *(const v16bf*)(krow + kbB);
    v16bf b1 = *(const v16bf*)(krow + 32 + kbB);
    v8f acc = {};
    acc = wmma_bf16(qa0, b0, acc);
    acc = wmma_bf16(qa1, b1, acc);
#pragma unroll
    for (int v = 0; v < 8; ++v)
      sm[(v + hi * 8) * T_SEQ + s0 + n] = acc[v];
  }
  __syncthreads();

  // ---- Phase 2: softmax (8 threads per row)
  const int m = tid >> 3;
  const int j = tid & 7;
  float mx = -3.4e38f;
  for (int c = j; c < T_SEQ; c += 8) mx = fmaxf(mx, sm[m * T_SEQ + c]);
  redmax[m * 8 + j] = mx;
  __syncthreads();
  float rowmax = redmax[m * 8];
#pragma unroll
  for (int q = 1; q < 8; ++q) rowmax = fmaxf(rowmax, redmax[m * 8 + q]);
  float sum = 0.f;
  for (int c = j; c < T_SEQ; c += 8) {
    float ex = __expf(sm[m * T_SEQ + c] - rowmax);
    sm[m * T_SEQ + c] = ex;
    sum += ex;
  }
  redsum[m * 8 + j] = sum;
  __syncthreads();
  if (tid < 16) {
    float s = 0.f;
#pragma unroll
    for (int q = 0; q < 8; ++q) s += redsum[tid * 8 + q];
    rs[tid] = 1.0f / s;
  }
  __syncthreads();
  // normalize + stream attn_weights (non-temporal: 536MB stream, spare L2)
  float* awrow = AW + (size_t)bh * T_SEQ * T_SEQ + (size_t)t0 * T_SEQ;
  for (int e = tid; e < 16 * T_SEQ; e += 128) {
    int mm = e >> 11;
    float p = sm[e] * rs[mm];
    sm[e] = p;
    __builtin_nontemporal_store(p, awrow + e);
  }
  __syncthreads();

  // ---- Phase 3: O = P.V  (wave -> 16-wide d tile)
  const int d0 = wave * 16;
  const bf16* vrow = Vt + (size_t)bh * DH * T_SEQ + (size_t)(d0 + n) * T_SEQ;
  v8f acc = {};
  for (int sc = 0; sc < T_SEQ; sc += 32) {
    const float* prow = sm + n * T_SEQ + sc;      // row = lane&15
    v16bf af = a_frag_from_f32(prow, kbA);        // LDS f32 -> bf16
    v16bf bf_ = *(const v16bf*)(vrow + sc + kbB); // contiguous along S
    acc = wmma_bf16(af, bf_, acc);
  }
  const int b_ = bh >> 4;
  const int h_ = bh & 15;
#pragma unroll
  for (int v = 0; v < 8; ++v) {
    int t = t0 + v + hi * 8;
    Ctx[(size_t)(t * BSZ + b_) * EMB + h_ * 64 + d0 + n] = (bf16)acc[v];
  }
}

// ---------------------------------------------------------------------------
// Kernel 3: output projection  out = Ctx(bf16) * Wo^T + bo, 4x4 blocked.
// ---------------------------------------------------------------------------
__global__ void __launch_bounds__(32, 1)
out_proj_kernel(const bf16* __restrict__ Ctx,
                const bf16* __restrict__ Wob,
                const float* __restrict__ bo,
                float* __restrict__ out) {
  const int lane = threadIdx.x;
  const int r0 = blockIdx.x * 64;
  const int o0 = blockIdx.y * 64;
  const int n   = lane & 15;
  const int hi  = lane >> 4;
  const int kbA = hi * 16;
  const int kbB = hi * 16;

  const bf16* cbase = Ctx + (size_t)(r0 + n) * EMB;
  const bf16* wbase = Wob + (size_t)(o0 + n) * EMB + kbB;

  v8f acc[4][4] = {};
  for (int e0 = 0; e0 < EMB; e0 += 32) {
    __builtin_prefetch(cbase + e0 + 256, 0, 3);
    __builtin_prefetch(wbase + e0 + 256, 0, 3);
    v16bf a[4], b[4];
#pragma unroll
    for (int mi = 0; mi < 4; ++mi)
      a[mi] = a_frag_from_bf16(cbase + mi * 16 * EMB + e0, kbA);
#pragma unroll
    for (int ni = 0; ni < 4; ++ni)
      b[ni] = *(const v16bf*)(wbase + ni * 16 * EMB + e0);
#pragma unroll
    for (int mi = 0; mi < 4; ++mi)
#pragma unroll
      for (int ni = 0; ni < 4; ++ni)
        acc[mi][ni] = wmma_bf16(a[mi], b[ni], acc[mi][ni]);
  }

#pragma unroll
  for (int mi = 0; mi < 4; ++mi)
#pragma unroll
    for (int ni = 0; ni < 4; ++ni) {
      const int o = o0 + ni * 16 + n;
      const float bb = bo[o];
#pragma unroll
      for (int v = 0; v < 8; ++v) {
        int r = r0 + mi * 16 + v + hi * 8;
        out[(size_t)r * EMB + o] = acc[mi][ni][v] + bb;
      }
    }
}

// ---------------------------------------------------------------------------
extern "C" void kernel_launch(void* const* d_in, const int* in_sizes, int n_in,
                              void* d_out, int out_size, void* d_ws, size_t ws_size,
                              hipStream_t stream) {
  const float* query = (const float*)d_in[0];
  // d_in[1] = key, d_in[2] = value : unused (self-attention uses query)
  const float* bq = (const float*)d_in[3];
  const float* bk = (const float*)d_in[4];
  const float* bv = (const float*)d_in[5];
  const float* Wq = (const float*)d_in[6];
  const float* Wk = (const float*)d_in[7];
  const float* Wv = (const float*)d_in[8];
  const float* Wo = (const float*)d_in[9];
  const float* bo = (const float*)d_in[10];

  float* out = (float*)d_out;                       // attn  [4096,1024] f32
  float* AW  = out + (size_t)ROWS * EMB;            // attn_weights [32,2048,2048]

  const size_t MB = (size_t)1024 * 1024;
  char* ws = (char*)d_ws;
  bf16* Qh  = (bf16*)(ws + 0 * MB);    // [32][2048][64] bf16   8MB
  bf16* Kh  = (bf16*)(ws + 8 * MB);    // [32][2048][64] bf16   8MB
  bf16* Vt  = (bf16*)(ws + 16 * MB);   // [32][64][2048] bf16   8MB (transposed)
  bf16* Ctx = (bf16*)(ws + 24 * MB);   // [4096][1024]   bf16   8MB
  bf16* Xb  = (bf16*)(ws + 32 * MB);   // [4096][1024]   bf16   8MB
  bf16* Wqb = (bf16*)(ws + 40 * MB);   // [1024][1024]   bf16   2MB
  bf16* Wkb = (bf16*)(ws + 42 * MB);
  bf16* Wvb = (bf16*)(ws + 44 * MB);
  bf16* Wob = (bf16*)(ws + 46 * MB);

  // one-time bf16 down-converts
  cvt_bf16_kernel<<<(ROWS * EMB) / 2048, 256, 0, stream>>>(query, Xb, ROWS * EMB);
  cvt_bf16_kernel<<<(EMB * EMB) / 2048, 256, 0, stream>>>(Wq, Wqb, EMB * EMB);
  cvt_bf16_kernel<<<(EMB * EMB) / 2048, 256, 0, stream>>>(Wk, Wkb, EMB * EMB);
  cvt_bf16_kernel<<<(EMB * EMB) / 2048, 256, 0, stream>>>(Wv, Wvb, EMB * EMB);
  cvt_bf16_kernel<<<(EMB * EMB) / 2048, 256, 0, stream>>>(Wo, Wob, EMB * EMB);

  qkv_proj_kernel<<<dim3(ROWS / 64, EMB / 64, 3), 32, 0, stream>>>(
      Xb, Wqb, Wkb, Wvb, bq, bk, bv, Qh, Kh, Vt);

  const int smem_bytes = 16 * T_SEQ * 4 + 128 * 4 + 128 * 4 + 16 * 4;
  hipFuncSetAttribute(reinterpret_cast<const void*>(attention_kernel),
                      hipFuncAttributeMaxDynamicSharedMemorySize, smem_bytes);
  attention_kernel<<<dim3(T_SEQ / 16, BH), 128, smem_bytes, stream>>>(
      Qh, Kh, Vt, AW, Ctx);

  out_proj_kernel<<<dim3(ROWS / 64, EMB / 64), 32, 0, stream>>>(Ctx, Wob, bo, out);
}
